// Attention_2637109920263
// MI455X (gfx1250) — compile-verified, matrix operand order checked by reading, not run on hardware
//
#include <hip/hip_runtime.h>

typedef __attribute__((ext_vector_type(16))) _Float16 v16h;
typedef __attribute__((ext_vector_type(8)))  _Float16 v8h;
typedef __attribute__((ext_vector_type(8)))  float    v8f;
typedef __attribute__((ext_vector_type(4)))  int      v4i;

#define B_   16
#define C_   256
#define N_   1024
#define NH_  4
#define D_   64
#define G_   8
#define CPG_ (C_ / G_)      // 32
#define EPS_ 1e-5f
#define BR   64             // query tile rows
#define BC   64             // key/value tile cols
#define LDH  72             // f16 LDS row stride: 144B, 16B-aligned, conflict-free
#define LDA  40             // GEMM LDS stride: 80B, 16B-aligned, conflict-free
#define LDS_S 65            // f32 score LDS stride

#define AS1 __attribute__((address_space(1)))
#define AS3 __attribute__((address_space(3)))

#if __has_builtin(__builtin_amdgcn_global_load_async_to_lds_b128)
#define HAVE_ASYNC 1
#else
#define HAVE_ASYNC 0
#endif

__device__ __forceinline__ void wait_async0() {
#if __has_builtin(__builtin_amdgcn_s_wait_asynccnt)
  __builtin_amdgcn_s_wait_asynccnt(0);
#else
  asm volatile("s_wait_asynccnt 0" ::: "memory");
#endif
}

// -------------------------------------------------------------------------
// f16 WMMA 16x32 fragment from row-major LDS tile [row][k], row stride `rs`
// halves (rs*2 must be a multiple of 16 bytes). Per ISA wave32 layout:
//   lane&15 = row (M for A / N for B), (lane>>4)*8 = K half-offset;
//   halves 0..7 = K khalf..khalf+7, halves 8..15 = K 16+khalf..16+khalf+7
// -> exactly two 16B LDS vector loads per lane.
// -------------------------------------------------------------------------
__device__ __forceinline__ v16h load_frag_row(const _Float16* p, int rs) {
  int lane  = threadIdx.x & 31;
  int r     = lane & 15;
  int khalf = (lane >> 4) << 3;
  const _Float16* row = p + r * rs;
  v8h lo = *(const v8h*)(row + khalf);
  v8h hi = *(const v8h*)(row + 16 + khalf);
  v16h out;
#pragma unroll
  for (int i = 0; i < 8; ++i) { out[i] = lo[i]; out[8 + i] = hi[i]; }
  return out;
}

// -------------------------------------------------------------------------
// Kernel 1: GroupNorm -> f16 activations
// -------------------------------------------------------------------------
__global__ __launch_bounds__(256) void groupnorm_kernel(
    const float* __restrict__ x, const float* __restrict__ w,
    const float* __restrict__ b, _Float16* __restrict__ h16) {
  __shared__ float red[256];
  __shared__ float s_mu, s_rstd;
  int bg = blockIdx.x;
  int bb = bg / G_;
  int g  = bg % G_;
  const float* base = x + ((size_t)bb * C_ + (size_t)g * CPG_) * N_;
  const int elems = CPG_ * N_;

  float s = 0.f, ss = 0.f;
  for (int i = threadIdx.x; i < elems; i += blockDim.x) {
    float v = base[i];
    s += v; ss += v * v;
  }
  red[threadIdx.x] = s;
  __syncthreads();
  for (int off = 128; off > 0; off >>= 1) {
    if ((int)threadIdx.x < off) red[threadIdx.x] += red[threadIdx.x + off];
    __syncthreads();
  }
  float tot = red[0];
  __syncthreads();
  red[threadIdx.x] = ss;
  __syncthreads();
  for (int off = 128; off > 0; off >>= 1) {
    if ((int)threadIdx.x < off) red[threadIdx.x] += red[threadIdx.x + off];
    __syncthreads();
  }
  float tot2 = red[0];
  if (threadIdx.x == 0) {
    float mu  = tot / (float)elems;
    float var = tot2 / (float)elems - mu * mu;
    s_mu = mu;
    s_rstd = rsqrtf(var + EPS_);
  }
  __syncthreads();
  float mu = s_mu, rstd = s_rstd;
  _Float16* hb = h16 + ((size_t)bb * C_ + (size_t)g * CPG_) * N_;
  for (int i = threadIdx.x; i < elems; i += blockDim.x) {
    int c = g * CPG_ + (i >> 10);
    float v = (base[i] - mu) * rstd * w[c] + b[c];
    hb[i] = (_Float16)v;
  }
}

// -------------------------------------------------------------------------
// Kernel 2: f32 -> f16 weight conversion
// -------------------------------------------------------------------------
__global__ void cvt_f16_kernel(const float* __restrict__ src,
                               _Float16* __restrict__ dst, int n) {
  int i = blockIdx.x * blockDim.x + threadIdx.x;
  if (i < n) dst[i] = (_Float16)src[i];
}

// -------------------------------------------------------------------------
// Kernel 3: QKV GEMM, block computes 64x64 tile of (3C x N) per batch.
// LDS-staged, all fragments read as 2x ds_load_b128.
// -------------------------------------------------------------------------
__global__ __launch_bounds__(256) void qkv_gemm_kernel(
    const _Float16* __restrict__ w16,   // (3C, C)
    const float* __restrict__ bias,     // (3C)
    const _Float16* __restrict__ h16,   // (B, C, N)
    _Float16* __restrict__ qkv16) {     // (B, 3C, N)
  __shared__ __align__(16) _Float16 As[64][LDA];  // [m][k]
  __shared__ __align__(16) _Float16 Bs[64][LDA];  // [n][k] (transposed)

  const int nblk = N_ / 64;             // 16
  const int oblk = (3 * C_) / 64;       // 12
  int bidx = blockIdx.x / (oblk * nblk);
  int rem  = blockIdx.x % (oblk * nblk);
  int m0 = (rem / nblk) * 64;
  int n0 = (rem % nblk) * 64;
  const _Float16* A  = w16 + (size_t)m0 * C_;
  const _Float16* Bb = h16 + (size_t)bidx * C_ * N_ + n0;

  int wave = threadIdx.x >> 5, lane = threadIdx.x & 31;
  int t0 = wave * 2, t1 = wave * 2 + 1;
  int t0i = (t0 >> 2) * 16, t0j = (t0 & 3) * 16;
  int t1i = (t1 >> 2) * 16, t1j = (t1 & 3) * 16;
  v8f acc0 = {}, acc1 = {};

  int am = threadIdx.x >> 2;            // 0..63
  int ak = (threadIdx.x & 3) * 8;       // 0,8,16,24
  int bc = threadIdx.x >> 3;            // 0..31
  int bn = (threadIdx.x & 7) * 8;       // 0..56

  for (int k0 = 0; k0 < C_; k0 += 32) {
    __syncthreads();
    *(v8h*)&As[am][ak] = *(const v8h*)(A + (size_t)am * C_ + k0 + ak);
    v8h bv = *(const v8h*)(Bb + (size_t)(k0 + bc) * N_ + bn);
#pragma unroll
    for (int i = 0; i < 8; ++i) Bs[bn + i][bc] = bv[i];
    if (k0 + 32 < C_) {
      __builtin_prefetch(A + (size_t)am * C_ + k0 + 32 + ak, 0, 1);
      __builtin_prefetch(Bb + (size_t)(k0 + 32 + bc) * N_ + bn, 0, 1);
    }
    __syncthreads();
    v16h a0 = load_frag_row(&As[t0i][0], LDA);
    v16h b0 = load_frag_row(&Bs[t0j][0], LDA);
    acc0 = __builtin_amdgcn_wmma_f32_16x16x32_f16(false, a0, false, b0,
                                                  (short)0, acc0, false, false);
    v16h a1 = load_frag_row(&As[t1i][0], LDA);
    v16h b1 = load_frag_row(&Bs[t1j][0], LDA);
    acc1 = __builtin_amdgcn_wmma_f32_16x16x32_f16(false, a1, false, b1,
                                                  (short)0, acc1, false, false);
  }

  int ncol = lane & 15;
  int mb   = (lane >> 4) << 3;
  _Float16* ob = qkv16 + (size_t)bidx * 3 * C_ * N_ + (size_t)m0 * N_ + n0;
#pragma unroll
  for (int r = 0; r < 8; ++r) {
    int m = t0i + mb + r;
    ob[(size_t)m * N_ + t0j + ncol] = (_Float16)(acc0[r] + bias[m0 + m]);
    int m2 = t1i + mb + r;
    ob[(size_t)m2 * N_ + t1j + ncol] = (_Float16)(acc1[r] + bias[m0 + m2]);
  }
}

// -------------------------------------------------------------------------
// Kernel 4: flash attention per (b, head, 64-row query tile).
// Qs/Ks transposed [n][c]; Vs natural [c][m] (async global->LDS when
// available). All WMMA fragments = 2x ds_load_b128.
// -------------------------------------------------------------------------
__global__ __launch_bounds__(256) void attn_kernel(
    const _Float16* __restrict__ qkv16,   // (B, 3C, N)
    _Float16* __restrict__ o16) {         // (B, C, N)
  __shared__ __align__(16) _Float16 Qs[BR][LDH];   // [i][c]
  __shared__ __align__(16) _Float16 Ks[BC][LDH];   // [j][c]
  __shared__ __align__(16) _Float16 Vs[D_][LDH];   // [c][m]
  __shared__ __align__(16) _Float16 Ps[BR][LDH];   // [i][m]
  __shared__ float Ss[BR][LDS_S];
  __shared__ float rowmax[BR], rowsum[BR], rowcorr[BR];
  __shared__ float pmax[4][BR], psum[4][BR];

  const int ntiles = N_ / BR;             // 16
  int bh = blockIdx.x / ntiles;
  int qt = blockIdx.x % ntiles;
  int bb = bh / NH_;
  int hh = bh % NH_;
  const _Float16* qb = qkv16 + ((size_t)bb * 3 + 0) * C_ * N_ + (size_t)hh * D_ * N_;
  const _Float16* kb = qkv16 + ((size_t)bb * 3 + 1) * C_ * N_ + (size_t)hh * D_ * N_;
  const _Float16* vb = qkv16 + ((size_t)bb * 3 + 2) * C_ * N_ + (size_t)hh * D_ * N_;
  int qn0 = qt * BR;

  // staging indices: each thread moves 16 halves (2x v8h)
  int sc = threadIdx.x >> 2;              // 0..63 : source row (channel)
  int si = (threadIdx.x & 3) << 4;        // 0,16,32,48 : column chunk

  { // Q tile (transpose c-major -> [i][c])
    v8h q0 = *(const v8h*)(qb + (size_t)sc * N_ + qn0 + si);
    v8h q1 = *(const v8h*)(qb + (size_t)sc * N_ + qn0 + si + 8);
#pragma unroll
    for (int j = 0; j < 8; ++j) { Qs[si + j][sc] = q0[j]; Qs[si + 8 + j][sc] = q1[j]; }
  }
  for (int i = threadIdx.x; i < BR; i += blockDim.x) {
    rowmax[i] = -1e30f; rowsum[i] = 0.f;
  }
  __syncthreads();

  int wave = threadIdx.x >> 5;
  int lane = threadIdx.x & 31;
  v8f oacc0 = {}; v8f oacc1 = {};
  int t0 = wave * 2, t1 = wave * 2 + 1;
  int t0i = (t0 >> 2) * 16, t0j = (t0 & 3) * 16;
  int t1i = (t1 >> 2) * 16, t1j = (t1 & 3) * 16;
  const float scale = 0.125f;             // d^-0.5, d=64

  for (int kt = 0; kt < N_ / BC; ++kt) {
    int kn0 = kt * BC;
    { // K tile: transpose into [j][c]
      v8h k0v = *(const v8h*)(kb + (size_t)sc * N_ + kn0 + si);
      v8h k1v = *(const v8h*)(kb + (size_t)sc * N_ + kn0 + si + 8);
#pragma unroll
      for (int j = 0; j < 8; ++j) { Ks[si + j][sc] = k0v[j]; Ks[si + 8 + j][sc] = k1v[j]; }
      // V tile: natural [c][m], straight 16B copies (async when available)
#if HAVE_ASYNC
      __builtin_amdgcn_global_load_async_to_lds_b128(
          (AS1 v4i*)(vb + (size_t)sc * N_ + kn0 + si),
          (AS3 v4i*)&Vs[sc][si], 0, 0);
      __builtin_amdgcn_global_load_async_to_lds_b128(
          (AS1 v4i*)(vb + (size_t)sc * N_ + kn0 + si + 8),
          (AS3 v4i*)&Vs[sc][si + 8], 0, 0);
#else
      *(v8h*)&Vs[sc][si]     = *(const v8h*)(vb + (size_t)sc * N_ + kn0 + si);
      *(v8h*)&Vs[sc][si + 8] = *(const v8h*)(vb + (size_t)sc * N_ + kn0 + si + 8);
#endif
    }
#if HAVE_ASYNC
    wait_async0();
#endif
    __syncthreads();

    // ---- S = (Q^T K) * scale ------------------------------------------
    {
      v8f s0 = {}; v8f s1 = {};
#pragma unroll
      for (int k0 = 0; k0 < D_; k0 += 32) {
        v16h a0 = load_frag_row(&Qs[t0i][k0], LDH);
        v16h b0 = load_frag_row(&Ks[t0j][k0], LDH);
        s0 = __builtin_amdgcn_wmma_f32_16x16x32_f16(false, a0, false, b0,
                                                    (short)0, s0, false, false);
        v16h a1 = load_frag_row(&Qs[t1i][k0], LDH);
        v16h b1 = load_frag_row(&Ks[t1j][k0], LDH);
        s1 = __builtin_amdgcn_wmma_f32_16x16x32_f16(false, a1, false, b1,
                                                    (short)0, s1, false, false);
      }
      int ncol = lane & 15;
      int mb   = (lane >> 4) << 3;
#pragma unroll
      for (int r = 0; r < 8; ++r) {
        Ss[t0i + mb + r][t0j + ncol] = s0[r] * scale;
        Ss[t1i + mb + r][t1j + ncol] = s1[r] * scale;
      }
    }
    __syncthreads();

    // ---- online softmax over this 64-col slab -------------------------
    {
      int row  = threadIdx.x & 63;
      int quad = threadIdx.x >> 6;
      float mx = -1e30f;
      for (int j = quad * 16; j < quad * 16 + 16; ++j) mx = fmaxf(mx, Ss[row][j]);
      pmax[quad][row] = mx;
      __syncthreads();
      if (quad == 0) {
        float m_new = fmaxf(fmaxf(pmax[0][row], pmax[1][row]),
                            fmaxf(pmax[2][row], pmax[3][row]));
        m_new = fmaxf(m_new, rowmax[row]);
        float corr = __expf(rowmax[row] - m_new);
        rowcorr[row] = corr;
        rowmax[row]  = m_new;
        rowsum[row] *= corr;
      }
      __syncthreads();
      float m_new = rowmax[row];
      float sm = 0.f;
      for (int j = quad * 16; j < quad * 16 + 16; ++j) {
        float p = __expf(Ss[row][j] - m_new);
        Ps[row][j] = (_Float16)p;
        sm += p;
      }
      psum[quad][row] = sm;
      __syncthreads();
      if (quad == 0)
        rowsum[row] += psum[0][row] + psum[1][row] + psum[2][row] + psum[3][row];
      __syncthreads();
    }

    // ---- O = O*corr + P V^T -------------------------------------------
    {
      int mb = (lane >> 4) << 3;
#pragma unroll
      for (int r = 0; r < 8; ++r) {
        oacc0[r] *= rowcorr[t0i + mb + r];
        oacc1[r] *= rowcorr[t1i + mb + r];
      }
#pragma unroll
      for (int k0 = 0; k0 < BC; k0 += 32) {
        v16h a0 = load_frag_row(&Ps[t0i][k0], LDH);   // A[m=i][k=m']
        v16h b0 = load_frag_row(&Vs[t0j][k0], LDH);   // B[k=m'][n=c] = Vs[c][m']
        oacc0 = __builtin_amdgcn_wmma_f32_16x16x32_f16(false, a0, false, b0,
                                                       (short)0, oacc0, false, false);
        v16h a1 = load_frag_row(&Ps[t1i][k0], LDH);
        v16h b1 = load_frag_row(&Vs[t1j][k0], LDH);
        oacc1 = __builtin_amdgcn_wmma_f32_16x16x32_f16(false, a1, false, b1,
                                                       (short)0, oacc1, false, false);
      }
    }
    __syncthreads();
  }

  // ---- finalize: divide by row sums, store O[b, h*64+c, qn0+i] --------
  {
    int ncol = lane & 15;
    int mb   = (lane >> 4) << 3;
    _Float16* ob = o16 + (size_t)bb * C_ * N_ + (size_t)hh * D_ * N_;
#pragma unroll
    for (int r = 0; r < 8; ++r) {
      int i0 = t0i + mb + r, c0 = t0j + ncol;
      ob[(size_t)c0 * N_ + qn0 + i0] = (_Float16)(oacc0[r] / rowsum[i0]);
      int i1 = t1i + mb + r, c1 = t1j + ncol;
      ob[(size_t)c1 * N_ + qn0 + i1] = (_Float16)(oacc1[r] / rowsum[i1]);
    }
  }
}

// -------------------------------------------------------------------------
// Kernel 5: proj GEMM + bias + residual, f32 out; 64x64 block tiles
// -------------------------------------------------------------------------
__global__ __launch_bounds__(256) void proj_gemm_kernel(
    const _Float16* __restrict__ w16,   // (C, C)
    const float* __restrict__ bias,     // (C)
    const _Float16* __restrict__ a16,   // (B, C, N)
    const float* __restrict__ x,        // (B, C, N)
    float* __restrict__ out) {
  __shared__ __align__(16) _Float16 As[64][LDA];
  __shared__ __align__(16) _Float16 Bs[64][LDA];

  const int nblk = N_ / 64;             // 16
  const int oblk = C_ / 64;             // 4
  int bidx = blockIdx.x / (oblk * nblk);
  int rem  = blockIdx.x % (oblk * nblk);
  int m0 = (rem / nblk) * 64;
  int n0 = (rem % nblk) * 64;
  const _Float16* A  = w16 + (size_t)m0 * C_;
  const _Float16* Bb = a16 + (size_t)bidx * C_ * N_ + n0;

  int wave = threadIdx.x >> 5, lane = threadIdx.x & 31;
  int t0 = wave * 2, t1 = wave * 2 + 1;
  int t0i = (t0 >> 2) * 16, t0j = (t0 & 3) * 16;
  int t1i = (t1 >> 2) * 16, t1j = (t1 & 3) * 16;
  v8f acc0 = {}, acc1 = {};

  int am = threadIdx.x >> 2;
  int ak = (threadIdx.x & 3) * 8;
  int bc = threadIdx.x >> 3;
  int bn = (threadIdx.x & 7) * 8;

  for (int k0 = 0; k0 < C_; k0 += 32) {
    __syncthreads();
    *(v8h*)&As[am][ak] = *(const v8h*)(A + (size_t)am * C_ + k0 + ak);
    v8h bv = *(const v8h*)(Bb + (size_t)(k0 + bc) * N_ + bn);
#pragma unroll
    for (int i = 0; i < 8; ++i) Bs[bn + i][bc] = bv[i];
    if (k0 + 32 < C_) {
      __builtin_prefetch(A + (size_t)am * C_ + k0 + 32 + ak, 0, 1);
      __builtin_prefetch(Bb + (size_t)(k0 + 32 + bc) * N_ + bn, 0, 1);
    }
    __syncthreads();
    v16h a0 = load_frag_row(&As[t0i][0], LDA);
    v16h b0 = load_frag_row(&Bs[t0j][0], LDA);
    acc0 = __builtin_amdgcn_wmma_f32_16x16x32_f16(false, a0, false, b0,
                                                  (short)0, acc0, false, false);
    v16h a1 = load_frag_row(&As[t1i][0], LDA);
    v16h b1 = load_frag_row(&Bs[t1j][0], LDA);
    acc1 = __builtin_amdgcn_wmma_f32_16x16x32_f16(false, a1, false, b1,
                                                  (short)0, acc1, false, false);
  }

  int ncol = lane & 15;
  int mb   = (lane >> 4) << 3;
  size_t base = (size_t)bidx * C_ * N_ + (size_t)m0 * N_ + n0;
#pragma unroll
  for (int r = 0; r < 8; ++r) {
    int m = t0i + mb + r;
    size_t idx0 = base + (size_t)m * N_ + t0j + ncol;
    out[idx0] = x[idx0] + bias[m0 + m] + acc0[r];
    int m2 = t1i + mb + r;
    size_t idx1 = base + (size_t)m2 * N_ + t1j + ncol;
    out[idx1] = x[idx1] + bias[m0 + m2] + acc1[r];
  }
}

// -------------------------------------------------------------------------
extern "C" void kernel_launch(void* const* d_in, const int* in_sizes, int n_in,
                              void* d_out, int out_size, void* d_ws, size_t ws_size,
                              hipStream_t stream) {
  const float* x      = (const float*)d_in[0];
  const float* norm_w = (const float*)d_in[1];
  const float* norm_b = (const float*)d_in[2];
  const float* qkv_w  = (const float*)d_in[3];
  const float* qkv_b  = (const float*)d_in[4];
  const float* proj_w = (const float*)d_in[5];
  const float* proj_b = (const float*)d_in[6];
  float* out = (float*)d_out;

  char* ws = (char*)d_ws;
  _Float16* h16     = (_Float16*)ws;  ws += (size_t)B_ * C_ * N_ * 2;
  _Float16* qkv16   = (_Float16*)ws;  ws += (size_t)B_ * 3 * C_ * N_ * 2;
  _Float16* attn16  = (_Float16*)ws;  ws += (size_t)B_ * C_ * N_ * 2;
  _Float16* qkvw16  = (_Float16*)ws;  ws += (size_t)3 * C_ * C_ * 2;
  _Float16* projw16 = (_Float16*)ws;  ws += (size_t)C_ * C_ * 2;
  (void)ws_size; (void)in_sizes; (void)n_in; (void)out_size;

  cvt_f16_kernel<<<(3 * C_ * C_ + 255) / 256, 256, 0, stream>>>(qkv_w, qkvw16, 3 * C_ * C_);
  cvt_f16_kernel<<<(C_ * C_ + 255) / 256, 256, 0, stream>>>(proj_w, projw16, C_ * C_);
  groupnorm_kernel<<<B_ * G_, 256, 0, stream>>>(x, norm_w, norm_b, h16);
  qkv_gemm_kernel<<<B_ * ((3 * C_) / 64) * (N_ / 64), 256, 0, stream>>>(qkvw16, qkv_b, h16, qkv16);
  attn_kernel<<<B_ * NH_ * (N_ / BR), 256, 0, stream>>>(qkv16, attn16);
  proj_gemm_kernel<<<B_ * (C_ / 64) * (N_ / 64), 256, 0, stream>>>(projw16, proj_b, attn16, x, out);
}
